// SelfAttentionLayer_47536698032950
// MI455X (gfx1250) — matrix operand order repaired and checked by reading, not compiled, not run on hardware
//
#include <hip/hip_runtime.h>
#include <hip/hip_bf16.h>

// ---------------- problem dims (compile-time) ----------------
#define BATCH    4
#define SEQ      1024
#define D_MODEL  1024
#define D_INNER  2048          // 2*D_MODEL
#define D_XZ     4096          // 2*D_INNER
#define DT_RANK  64
#define D_STATE  16
#define XPROJ    96            // DT_RANK + 2*D_STATE
#define D_CONV   4
#define ROWS     (BATCH*SEQ)   // 4096 token rows

typedef __attribute__((ext_vector_type(8)))  __bf16 v8bf;
typedef __attribute__((ext_vector_type(16))) __bf16 v16bf;
typedef __attribute__((ext_vector_type(8)))  float  v8f;

// ---------------- helpers ----------------
__device__ __forceinline__ __bf16 f2bf(float f) {
  unsigned u = __builtin_bit_cast(unsigned, f);
  u += 0x7FFFu + ((u >> 16) & 1u);            // round-to-nearest-even
  unsigned short h = (unsigned short)(u >> 16);
  return __builtin_bit_cast(__bf16, h);
}
__device__ __forceinline__ float silu_f(float x) {
  return x / (1.f + __expf(-x));
}

// A-fragment (16x32 bf16): lane m = m0+lane%16, half=lane/16.
// element e<8 -> K = k + half*8 + e ; e>=8 -> K = k + 16 + half*8 + (e-8)
__device__ __forceinline__ v16bf load_a_frag(const __bf16* p /* row + k + half*8 */) {
  v8bf lo = *(const v8bf*)p;
  v8bf hi = *(const v8bf*)(p + 16);
  v16bf r;
#pragma unroll
  for (int i = 0; i < 8; ++i) { r[i] = lo[i]; r[i + 8] = hi[i]; }
  return r;
}
// B-fragment (32x16 bf16): lane n = n0+lane%16; element e<8 -> K = k + half*8 + e ; e>=8 -> K = k + 16 + half*8 + (e-8)
__device__ __forceinline__ v16bf load_b_frag(const __bf16* p /* row(n) + k + half*8 */) {
  v8bf lo = *(const v8bf*)p;
  v8bf hi = *(const v8bf*)(p + 16);
  v16bf r;
#pragma unroll
  for (int i = 0; i < 8; ++i) { r[i] = lo[i]; r[i + 8] = hi[i]; }
  return r;
}

// ---------------- tiny kernels ----------------
__global__ void k_silu_vec(const float* __restrict__ x, float* __restrict__ y, int n) {
  int i = blockIdx.x * blockDim.x + threadIdx.x;
  if (i < n) y[i] = silu_f(x[i]);
}

// mod[b][e] = sum_d silu_ts[b][d]*adaln_w[e][d] + adaln_b[e]   (e in [0,2048))
__global__ void k_adaln_mod(const float* __restrict__ s, const float* __restrict__ w,
                            const float* __restrict__ bias, float* __restrict__ mod) {
  int e = blockIdx.x * blockDim.x + threadIdx.x;
  int b = blockIdx.y;
  if (e >= 2 * D_MODEL) return;
  const float* sr = s + b * D_MODEL;
  const float* wr = w + (size_t)e * D_MODEL;
  float acc = bias[e];
#pragma unroll 4
  for (int d = 0; d < D_MODEL; ++d) acc += sr[d] * wr[d];
  mod[b * 2 * D_MODEL + e] = acc;
}

// aq = q*(1+scale)+shift  -> bf16
__global__ void k_modulate_bf16(const float* __restrict__ q, const float* __restrict__ mod,
                                __bf16* __restrict__ aq) {
  size_t i = (size_t)blockIdx.x * blockDim.x + threadIdx.x;
  if (i >= (size_t)ROWS * D_MODEL) return;
  int dm = (int)(i % D_MODEL);
  int b  = (int)(i / ((size_t)SEQ * D_MODEL));
  float scale = mod[b * 2 * D_MODEL + dm];
  float shift = mod[b * 2 * D_MODEL + D_MODEL + dm];
  aq[i] = f2bf(q[i] * (1.f + scale) + shift);
}

__global__ void k_f32_to_bf16(const float* __restrict__ in, __bf16* __restrict__ out, size_t n) {
  size_t i = (size_t)blockIdx.x * blockDim.x + threadIdx.x;
  if (i < n) out[i] = f2bf(in[i]);
}

// ---------------- WMMA GEMM: C[M,N] = A[M,K] * Bt[N,K]^T (+resid), bf16 in / f32 out ----
// One wave computes a 64x64 tile as 4x4 WMMA accumulators; K stepped by 32.
template <bool HAS_RESID>
__global__ __launch_bounds__(32) void k_gemm_bf16_wmma(
    const __bf16* __restrict__ A, const __bf16* __restrict__ Bt,
    const float* __restrict__ resid, float* __restrict__ C,
    int M, int N, int K) {
  const int lane = threadIdx.x & 31;
  const int m16  = lane & 15;
  const int half = lane >> 4;
  const int n0 = blockIdx.x * 64;
  const int m0 = blockIdx.y * 64;

  const v8f vz = {0.f,0.f,0.f,0.f,0.f,0.f,0.f,0.f};
  v8f acc[4][4];
#pragma unroll
  for (int mi = 0; mi < 4; ++mi)
#pragma unroll
    for (int ni = 0; ni < 4; ++ni) acc[mi][ni] = vz;

  const __bf16* Ap = A + (size_t)(m0 + m16) * K + half * 8;
  const __bf16* Bp = Bt + (size_t)(n0 + m16) * K + half * 8;

  for (int k = 0; k < K; k += 32) {
    v16bf af[4], bfr[4];
#pragma unroll
    for (int t = 0; t < 4; ++t)
      af[t] = load_a_frag(Ap + (size_t)(t * 16) * K + k);
#pragma unroll
    for (int t = 0; t < 4; ++t)
      bfr[t] = load_b_frag(Bp + (size_t)(t * 16) * K + k);
#pragma unroll
    for (int mi = 0; mi < 4; ++mi)
#pragma unroll
      for (int ni = 0; ni < 4; ++ni)
        acc[mi][ni] = __builtin_amdgcn_wmma_f32_16x16x32_bf16(
            false, af[mi], false, bfr[ni], (short)0, acc[mi][ni], false, false);
  }

  // D layout: vgpr r -> m = m0+mi*16 + half*8 + r ; n = n0+ni*16 + lane%16
#pragma unroll
  for (int mi = 0; mi < 4; ++mi) {
#pragma unroll
    for (int r = 0; r < 8; ++r) {
      const int m = m0 + mi * 16 + half * 8 + r;
      float* crow = C + (size_t)m * N + n0 + m16;
      const float* rrow = HAS_RESID ? (resid + (size_t)m * N + n0 + m16) : nullptr;
#pragma unroll
      for (int ni = 0; ni < 4; ++ni) {
        float v = acc[mi][ni][r];
        if (HAS_RESID) v += rrow[ni * 16];
        crow[ni * 16] = v;
      }
    }
  }
}

// ---------------- conv1d(d_conv=4, causal) + SiLU ----------------
__global__ void k_conv_silu(const float* __restrict__ xz, const float* __restrict__ cw,
                            const float* __restrict__ cb, float* __restrict__ u_act) {
  size_t idx = (size_t)blockIdx.x * blockDim.x + threadIdx.x;
  if (idx >= (size_t)ROWS * D_INNER) return;
  int d = (int)(idx % D_INNER);
  size_t bl = idx / D_INNER;
  int l = (int)(bl % SEQ);
  int b = (int)(bl / SEQ);
  float acc = cb[d];
#pragma unroll
  for (int k = 0; k < D_CONV; ++k) {
    int ls = l - (D_CONV - 1) + k;
    if (ls >= 0)
      acc += xz[((size_t)b * SEQ + ls) * D_XZ + d] * cw[d * D_CONV + k];
  }
  u_act[idx] = silu_f(acc);
}

// ---------------- x_dbl[i][j] = sum_d u_act[i][d] * w_xproj[j][d]  (j<96) -------------
__global__ __launch_bounds__(128) void k_xproj(const float* __restrict__ u_act,
                                               const float* __restrict__ w_xproj,
                                               float* __restrict__ x_dbl) {
  __shared__ float row[D_INNER];
  int i = blockIdx.x;
  int tid = threadIdx.x;
  const float* ur = u_act + (size_t)i * D_INNER;
  for (int j = tid; j < D_INNER; j += 128) row[j] = ur[j];
  __syncthreads();
  if (tid < XPROJ) {
    const float* wr = w_xproj + (size_t)tid * D_INNER;
    float acc = 0.f;
#pragma unroll 4
    for (int d = 0; d < D_INNER; ++d) acc += row[d] * wr[d];
    x_dbl[(size_t)i * XPROJ + tid] = acc;
  }
}

// ---------------- delta = softplus(dt @ w_dt^T + b_dt) ----------------
__global__ void k_delta(const float* __restrict__ x_dbl, const float* __restrict__ w_dt,
                        const float* __restrict__ b_dt, float* __restrict__ delta) {
  size_t idx = (size_t)blockIdx.x * blockDim.x + threadIdx.x;
  if (idx >= (size_t)ROWS * D_INNER) return;
  int d = (int)(idx & (D_INNER - 1));
  size_t i = idx >> 11;                       // / D_INNER
  const float* xr = x_dbl + i * XPROJ;        // dt part = first DT_RANK
  const float* wr = w_dt + (size_t)d * DT_RANK;
  float acc = b_dt[d];
#pragma unroll 8
  for (int r = 0; r < DT_RANK; ++r) acc += xr[r] * wr[r];
  delta[idx] = (acc > 20.f) ? acc : log1pf(__expf(acc));
}

// ---------------- selective scan (serial in L) + gate, bf16 out ----------------
__global__ __launch_bounds__(256) void k_scan(
    const float* __restrict__ delta, const float* __restrict__ u_act,
    const float* __restrict__ x_dbl, const float* __restrict__ xz,
    const float* __restrict__ a_log, const float* __restrict__ d_skip,
    __bf16* __restrict__ y_out) {
  int d = blockIdx.x * blockDim.x + threadIdx.x;   // channel
  int b = blockIdx.y;
  if (d >= D_INNER) return;
  float Ad[D_STATE], h[D_STATE];
#pragma unroll
  for (int s = 0; s < D_STATE; ++s) {
    Ad[s] = -__expf(a_log[(size_t)d * D_STATE + s]);
    h[s] = 0.f;
  }
  const float dsk = d_skip[d];
  for (int l = 0; l < SEQ; ++l) {
    size_t row = (size_t)b * SEQ + l;
    float dlt = delta[row * D_INNER + d];
    float ut  = u_act[row * D_INNER + d];
    const float* bc = x_dbl + row * XPROJ + DT_RANK;   // B then C, 16 each
    float du = dlt * ut;
    float y = 0.f;
#pragma unroll
    for (int s = 0; s < D_STATE; ++s) {
      float dA = __expf(dlt * Ad[s]);
      h[s] = dA * h[s] + du * bc[s];
      y += h[s] * bc[D_STATE + s];
    }
    float zt = xz[row * D_XZ + D_INNER + d];
    y_out[row * D_INNER + d] = f2bf((y + ut * dsk) * silu_f(zt));
  }
}

// ---------------- layernorm over last dim (1024) ----------------
__global__ __launch_bounds__(256) void k_layernorm(const float* __restrict__ x,
                                                   const float* __restrict__ g,
                                                   const float* __restrict__ bta,
                                                   float* __restrict__ out) {
  __shared__ float red[256];
  int row = blockIdx.x, tid = threadIdx.x;
  const float* xr = x + (size_t)row * D_MODEL;
  float s = 0.f, s2 = 0.f;
  for (int j = tid; j < D_MODEL; j += 256) { float v = xr[j]; s += v; s2 += v * v; }
  red[tid] = s; __syncthreads();
  for (int o = 128; o > 0; o >>= 1) { if (tid < o) red[tid] += red[tid + o]; __syncthreads(); }
  float mean = red[0] / (float)D_MODEL;
  __syncthreads();
  red[tid] = s2; __syncthreads();
  for (int o = 128; o > 0; o >>= 1) { if (tid < o) red[tid] += red[tid + o]; __syncthreads(); }
  float var = red[0] / (float)D_MODEL - mean * mean;
  float inv = rsqrtf(var + 1e-5f);
  for (int j = tid; j < D_MODEL; j += 256) {
    float v = (xr[j] - mean) * inv;
    out[(size_t)row * D_MODEL + j] = v * g[j] + bta[j];
  }
}

// ---------------- launch ----------------
extern "C" void kernel_launch(void* const* d_in, const int* in_sizes, int n_in,
                              void* d_out, int out_size, void* d_ws, size_t ws_size,
                              hipStream_t stream) {
  const float* q       = (const float*)d_in[0];
  const float* diff_ts = (const float*)d_in[1];
  const float* adaln_w = (const float*)d_in[2];
  const float* adaln_b = (const float*)d_in[3];
  const float* w_in    = (const float*)d_in[4];
  const float* conv_w  = (const float*)d_in[5];
  const float* conv_b  = (const float*)d_in[6];
  const float* w_xproj = (const float*)d_in[7];
  const float* w_dt    = (const float*)d_in[8];
  const float* b_dt    = (const float*)d_in[9];
  const float* a_log   = (const float*)d_in[10];
  const float* d_skip  = (const float*)d_in[11];
  const float* w_out   = (const float*)d_in[12];
  const float* ln_g    = (const float*)d_in[13];
  const float* ln_b    = (const float*)d_in[14];
  float* out = (float*)d_out;

  char* ws = (char*)d_ws;
  size_t off = 0;
  auto alloc = [&](size_t bytes) { char* p = ws + off; off += (bytes + 255) & ~(size_t)255; return p; };
  __bf16* w_in_bf  = (__bf16*)alloc((size_t)D_XZ * D_MODEL * 2);      // 8 MB
  __bf16* w_out_bf = (__bf16*)alloc((size_t)D_MODEL * D_INNER * 2);   // 4 MB
  __bf16* aq_bf    = (__bf16*)alloc((size_t)ROWS * D_MODEL * 2);      // 8 MB
  float*  xz       = (float*)alloc((size_t)ROWS * D_XZ * 4);          // 64 MB
  float*  u_act    = (float*)alloc((size_t)ROWS * D_INNER * 4);       // 32 MB
  float*  x_dbl    = (float*)alloc((size_t)ROWS * XPROJ * 4);         // 1.5 MB
  float*  delta    = (float*)alloc((size_t)ROWS * D_INNER * 4);       // 32 MB
  __bf16* y_bf     = (__bf16*)alloc((size_t)ROWS * D_INNER * 2);      // 16 MB
  float*  silu_ts  = (float*)alloc((size_t)BATCH * D_MODEL * 4);
  float*  mod      = (float*)alloc((size_t)BATCH * 2 * D_MODEL * 4);
  float*  out_pre  = xz;   // xz is dead after the scan; reuse for residual-add output

  // 1) AdaLN modulation
  k_silu_vec<<<(BATCH * D_MODEL + 255) / 256, 256, 0, stream>>>(diff_ts, silu_ts, BATCH * D_MODEL);
  k_adaln_mod<<<dim3((2 * D_MODEL + 255) / 256, BATCH), 256, 0, stream>>>(silu_ts, adaln_w, adaln_b, mod);
  k_modulate_bf16<<<(ROWS * D_MODEL + 255) / 256, 256, 0, stream>>>(q, mod, aq_bf);

  // 2) weight conversions to bf16
  k_f32_to_bf16<<<((size_t)D_XZ * D_MODEL + 255) / 256, 256, 0, stream>>>(w_in, w_in_bf, (size_t)D_XZ * D_MODEL);
  k_f32_to_bf16<<<((size_t)D_MODEL * D_INNER + 255) / 256, 256, 0, stream>>>(w_out, w_out_bf, (size_t)D_MODEL * D_INNER);

  // 3) in-projection GEMM: xz[4096,4096] = aq[4096,1024] @ w_in^T   (WMMA bf16)
  k_gemm_bf16_wmma<false><<<dim3(D_XZ / 64, ROWS / 64), 32, 0, stream>>>(
      aq_bf, w_in_bf, nullptr, xz, ROWS, D_XZ, D_MODEL);

  // 4) causal conv4 + SiLU
  k_conv_silu<<<((size_t)ROWS * D_INNER + 255) / 256, 256, 0, stream>>>(xz, conv_w, conv_b, u_act);

  // 5) x-projection (N=96, plain f32)
  k_xproj<<<ROWS, 128, 0, stream>>>(u_act, w_xproj, x_dbl);

  // 6) delta = softplus(dt-proj)
  k_delta<<<((size_t)ROWS * D_INNER + 255) / 256, 256, 0, stream>>>(x_dbl, w_dt, b_dt, delta);

  // 7) selective scan + skip + gate -> bf16
  k_scan<<<dim3(D_INNER / 256, BATCH), 256, 0, stream>>>(delta, u_act, x_dbl, xz, a_log, d_skip, y_bf);

  // 8) out-projection GEMM + residual: out_pre = q + y @ w_out^T   (WMMA bf16)
  k_gemm_bf16_wmma<true><<<dim3(D_MODEL / 64, ROWS / 64), 32, 0, stream>>>(
      y_bf, w_out_bf, q, out_pre, ROWS, D_MODEL, D_INNER);

  // 9) LayerNorm
  k_layernorm<<<ROWS, 256, 0, stream>>>(out_pre, ln_g, ln_b, out);
}
